// SeqNet_66073776882278
// MI455X (gfx1250) — matrix operand order repaired and checked by reading, not compiled, not run on hardware
//
#include <hip/hip_runtime.h>

// ---------------- problem constants (from reference) ----------------
#define TT   100
#define BB   512
#define FF   784
#define HH   1024
#define OO   10
#define KP   800          // FF padded to multiple of 32 for 16x16x32 WMMA
#define ROWS (TT*BB)      // 51200 rows of the batched GEMM

typedef __attribute__((ext_vector_type(16))) _Float16 v16h;
typedef __attribute__((ext_vector_type(8)))  float    v8f;

union Frag16 { v16h v; uint4 u[2]; };

// ---------------- cast + pad fp32 -> f16 ----------------
__global__ void cvt_pad_kernel(const float* __restrict__ src, _Float16* __restrict__ dst,
                               int rows, int scols, int dcols) {
    size_t idx = (size_t)blockIdx.x * blockDim.x + threadIdx.x;
    size_t total = (size_t)rows * dcols;
    if (idx >= total) return;
    int r = (int)(idx / dcols);
    int c = (int)(idx % dcols);
    dst[idx] = (c < scols) ? (_Float16)src[(size_t)r * scols + c] : (_Float16)0.0f;
}

// ---------------- batched input GEMM: zin[row,h] = x[row,:].W0[h,:] + b0[h] ----------
// Block = 256 threads (8 waves) computes a 128(M) x 128(N) macro-tile.
// Double-buffered LDS staging of A (128x32) and B (128x32) panels per k-step.
// Waves arranged 4(M) x 2(N); each wave computes 32x64 = 8 16x16 accumulators.
//
// WMMA fragment layouts (wave32, v_wmma_f32_16x16x32_f16):
//   A (16x32 f16): lane l -> row (l&15), K = 8*(l>>4) + {0..7} and {16..23}
//   B (32x16 f16): lane l -> col (l&15), K = 8*(l>>4) + {0..7} and {16..23}
//   D (16x16 f32): lane l, vgpr r -> M = r + 8*(l>>4), N = (l&15)
__global__ void __launch_bounds__(256)
gemm_zin_kernel(const _Float16* __restrict__ xh,
                const _Float16* __restrict__ w0h,
                const float*    __restrict__ b0,
                _Float16*       __restrict__ zin) {
    __shared__ _Float16 sA[2][128 * 32];
    __shared__ _Float16 sB[2][128 * 32];

    const int tid  = threadIdx.x;
    const int lane = tid & 31;
    const int wid  = tid >> 5;        // 0..7
    const int wm   = wid >> 1;        // 0..3 : 32-row band
    const int wn   = wid & 1;         // 0..1 : 64-col band

    const int mbase = blockIdx.x * 128;           // 400 M-tiles
    const int nbase = blockIdx.y * 128;           // 8  N-tiles

    // staging coords: thread loads 2 rows x 8 halves (16B) for each of A and B
    const int sr   = tid >> 2;                    // 0..63
    const int soff = (tid & 3) * 8;               // 0,8,16,24 halves

    v8f acc[2][4];
    #pragma unroll
    for (int sm = 0; sm < 2; ++sm)
        #pragma unroll
        for (int s = 0; s < 4; ++s)
            acc[sm][s] = (v8f){0.f,0.f,0.f,0.f,0.f,0.f,0.f,0.f};

    // stage k=0 into buffer 0
    {
        const _Float16* ga0 = xh  + (size_t)(mbase + sr)      * KP + soff;
        const _Float16* ga1 = xh  + (size_t)(mbase + sr + 64) * KP + soff;
        const _Float16* gb0 = w0h + (size_t)(nbase + sr)      * KP + soff;
        const _Float16* gb1 = w0h + (size_t)(nbase + sr + 64) * KP + soff;
        *(uint4*)&sA[0][sr * 32 + soff]        = *(const uint4*)ga0;
        *(uint4*)&sA[0][(sr + 64) * 32 + soff] = *(const uint4*)ga1;
        *(uint4*)&sB[0][sr * 32 + soff]        = *(const uint4*)gb0;
        *(uint4*)&sB[0][(sr + 64) * 32 + soff] = *(const uint4*)gb1;
    }

    const int NK = KP / 32;   // 25 k-steps
    for (int kk = 0; kk < NK; ++kk) {
        __syncthreads();
        int buf = kk & 1;
        // prefetch next panel into the other buffer
        if (kk + 1 < NK) {
            int kn = (kk + 1) * 32;
            const _Float16* ga0 = xh  + (size_t)(mbase + sr)      * KP + kn + soff;
            const _Float16* ga1 = xh  + (size_t)(mbase + sr + 64) * KP + kn + soff;
            const _Float16* gb0 = w0h + (size_t)(nbase + sr)      * KP + kn + soff;
            const _Float16* gb1 = w0h + (size_t)(nbase + sr + 64) * KP + kn + soff;
            int nb = buf ^ 1;
            *(uint4*)&sA[nb][sr * 32 + soff]        = *(const uint4*)ga0;
            *(uint4*)&sA[nb][(sr + 64) * 32 + soff] = *(const uint4*)ga1;
            *(uint4*)&sB[nb][sr * 32 + soff]        = *(const uint4*)gb0;
            *(uint4*)&sB[nb][(sr + 64) * 32 + soff] = *(const uint4*)gb1;
        }

        // A fragments for this wave's two 16-row sub-bands
        Frag16 a[2];
        #pragma unroll
        for (int sm = 0; sm < 2; ++sm) {
            const _Float16* p = &sA[buf][(wm * 32 + sm * 16 + (lane & 15)) * 32
                                         + (lane >> 4) * 8];
            a[sm].u[0] = *(const uint4*)(p);
            a[sm].u[1] = *(const uint4*)(p + 16);
        }
        // 4 B fragments, each fed to both A sub-bands
        #pragma unroll
        for (int s = 0; s < 4; ++s) {
            const _Float16* p = &sB[buf][(wn * 64 + s * 16 + (lane & 15)) * 32
                                         + (lane >> 4) * 8];
            Frag16 b;
            b.u[0] = *(const uint4*)(p);
            b.u[1] = *(const uint4*)(p + 16);
            #pragma unroll
            for (int sm = 0; sm < 2; ++sm)
                acc[sm][s] = __builtin_amdgcn_wmma_f32_16x16x32_f16(
                    false, a[sm].v, false, b.v, (short)0, acc[sm][s], false, false);
        }
    }

    // epilogue: add bias, convert to f16, store
    #pragma unroll
    for (int sm = 0; sm < 2; ++sm) {
        #pragma unroll
        for (int s = 0; s < 4; ++s) {
            int col = nbase + wn * 64 + s * 16 + (lane & 15);
            float bias = b0[col];
            #pragma unroll
            for (int r = 0; r < 8; ++r) {
                int m = mbase + wm * 32 + sm * 16 + r + 8 * (lane >> 4);
                zin[(size_t)m * HH + col] = (_Float16)(acc[sm][s][r] + bias);
            }
        }
    }
}

// ---------------- LIF recurrence + partial output contributions ----------------
// grid (4, 512): blockIdx.x = h-chunk of 256, blockIdx.y = batch b. Thread owns (b,h).
// S[t,b,o] += sum_{h in chunk} z[t,b,h] * Wout[o,h]   (wave-reduced, lane0 atomicAdd)
__global__ void lif_kernel(const _Float16* __restrict__ zin,
                           const float*    __restrict__ wout,
                           float*          __restrict__ S) {
    int b = blockIdx.y;
    int h = blockIdx.x * 256 + threadIdx.x;
    int lane = threadIdx.x & 31;

    float w[OO];
    #pragma unroll
    for (int o = 0; o < OO; ++o) w[o] = wout[(size_t)o * HH + h];

    float v = 0.f, cur = 0.f;
    for (int t = 0; t < TT; ++t) {
        float zi   = (float)zin[((size_t)(t * BB + b)) * HH + h];
        float vdec = 0.9f * v + 0.1f * cur;   // v + dt*tau_mem_inv*((0-v)+i)
        float idec = 0.8f * cur;              // i * (1 - dt*tau_syn_inv)
        bool  zsp  = (vdec - 1.0f) > 0.0f;    // superspike forward
        v   = zsp ? 0.0f : vdec;              // reset to V_RESET=0
        cur = idec + zi;

        #pragma unroll
        for (int o = 0; o < OO; ++o) {
            float s = zsp ? w[o] : 0.0f;
            s += __shfl_xor(s, 16, 32);
            s += __shfl_xor(s,  8, 32);
            s += __shfl_xor(s,  4, 32);
            s += __shfl_xor(s,  2, 32);
            s += __shfl_xor(s,  1, 32);
            if (lane == 0 && s != 0.0f)
                atomicAdd(&S[(size_t)t * (BB * OO) + b * OO + o], s);
        }
    }
}

// ---------------- LI (leaky integrator) output recurrence ----------------
__global__ void li_kernel(const float* __restrict__ S, float* __restrict__ out) {
    int idx = blockIdx.x * 256 + threadIdx.x;
    if (idx >= BB * OO) return;
    float v = 0.f, cur = 0.f;
    for (int t = 0; t < TT; ++t) {
        float s  = S[(size_t)t * (BB * OO) + idx];
        float vn = 0.9f * v + 0.1f * cur;
        cur = 0.8f * cur + s;
        v = vn;
        out[(size_t)t * (BB * OO) + idx] = vn;
    }
}

// ---------------- launch ----------------
extern "C" void kernel_launch(void* const* d_in, const int* in_sizes, int n_in,
                              void* d_out, int out_size, void* d_ws, size_t ws_size,
                              hipStream_t stream) {
    const float* x    = (const float*)d_in[0];   // [T,B,F]
    const float* W0   = (const float*)d_in[1];   // [H,F]
    const float* b0   = (const float*)d_in[2];   // [H]
    const float* Wout = (const float*)d_in[3];   // [O,H]
    float* out = (float*)d_out;                  // [T,B,O]

    const size_t XH_BYTES  = (size_t)ROWS * KP * sizeof(_Float16);   // 81,920,000
    const size_t W0H_BYTES = (size_t)HH   * KP * sizeof(_Float16);   //  1,638,400
    const size_t ZIN_BYTES = (size_t)ROWS * HH * sizeof(_Float16);   // 104,857,600
    const size_t S_BYTES   = (size_t)TT * BB * OO * sizeof(float);   //  2,048,000

    char* ws = (char*)d_ws;
    _Float16* xh  = (_Float16*)(ws);
    _Float16* w0h = (_Float16*)(ws + XH_BYTES);
    _Float16* zin = (_Float16*)(ws + XH_BYTES + W0H_BYTES);
    float*    S   = (float*)   (ws + XH_BYTES + W0H_BYTES + ZIN_BYTES);
    (void)ws_size; (void)in_sizes; (void)n_in; (void)out_size;

    // 1) cast + pad inputs to f16
    {
        size_t total = (size_t)ROWS * KP;
        cvt_pad_kernel<<<(int)((total + 255) / 256), 256, 0, stream>>>(x, xh, ROWS, FF, KP);
    }
    {
        size_t total = (size_t)HH * KP;
        cvt_pad_kernel<<<(int)((total + 255) / 256), 256, 0, stream>>>(W0, w0h, HH, FF, KP);
    }

    // 2) zero the partial-sum accumulator (atomics target)
    hipMemsetAsync(S, 0, S_BYTES, stream);

    // 3) batched input GEMM via WMMA: 400 x 8 macro-tiles of 128x128
    {
        dim3 grid(ROWS / 128, HH / 128);
        gemm_zin_kernel<<<grid, 256, 0, stream>>>(xh, w0h, b0, zin);
    }

    // 4) LIF recurrence + spike projection partials
    {
        dim3 grid(HH / 256, BB);
        lif_kernel<<<grid, 256, 0, stream>>>(zin, Wout, S);
    }

    // 5) LI output recurrence
    {
        int blocks = (BB * OO + 255) / 256;
        li_kernel<<<blocks, 256, 0, stream>>>(S, out);
    }
}
